// SingleHeaded_32246614459254
// MI455X (gfx1250) — matrix operand order repaired and checked by reading, not compiled, not run on hardware
//
#include <hip/hip_runtime.h>

typedef __attribute__((ext_vector_type(16))) __bf16 v16bf;
typedef __attribute__((ext_vector_type(8)))  __bf16 v8bf;
typedef __attribute__((ext_vector_type(8)))  float  v8f;
typedef __attribute__((ext_vector_type(4)))  unsigned int v4u;
typedef __attribute__((ext_vector_type(8)))  int v8i;
typedef __attribute__((ext_vector_type(4)))  int v4i;

#define TT 4096
#define CC 256
#define BB 4
#define MTOT (BB * TT) /* 16384 rows total */

#if defined(__HIP_DEVICE_COMPILE__) && defined(__gfx1250__)
#if __has_builtin(__builtin_amdgcn_tensor_load_to_lds)
#define HAVE_TDM 1
#endif
#if __has_builtin(__builtin_amdgcn_s_wait_tensorcnt)
#define HAVE_TCNT 1
#endif
#endif

static __device__ __forceinline__ v16bf mk16(v8bf lo, v8bf hi) {
  v16bf r;
#pragma unroll
  for (int i = 0; i < 8; ++i) { r[i] = lo[i]; r[i + 8] = hi[i]; }
  return r;
}

static __device__ __forceinline__ v8f wmma_bf16(v16bf a, v16bf b, v8f c) {
  // D = A(16x32) * B(32x16) + C(16x16 f32)
  return __builtin_amdgcn_wmma_f32_16x16x32_bf16(false, a, false, b,
                                                 (short)0, c, false, false);
}

static __device__ __forceinline__ float redmax16(float v) {
#pragma unroll
  for (int m = 1; m < 16; m <<= 1) v = fmaxf(v, __shfl_xor(v, m, 32));
  return v;
}
static __device__ __forceinline__ float redsum16(float v) {
#pragma unroll
  for (int m = 1; m < 16; m <<= 1) v += __shfl_xor(v, m, 32);
  return v;
}

// 32-bit LDS byte offset of a __shared__ object (generic addr low 32 bits)
static __device__ __forceinline__ unsigned lds_off(const void* p) {
  return (unsigned)(unsigned long long)p;
}

#if HAVE_TDM
// TDM: DMA a 32x256 bf16 tile (row stride 256) from global into LDS with
// 16B padding every 512B of payload -> matches Ks row stride of 264 elems.
static __device__ __forceinline__ void tdm_load_k_tile(const __bf16* gsrc,
                                                       unsigned lds_addr) {
  const unsigned long long ga = (unsigned long long)gsrc;
  v4u g0;
  g0[0] = 1u;                                   // count=1 (valid user D#)
  g0[1] = lds_addr;                             // lds_addr [63:32]
  g0[2] = (unsigned)(ga & 0xffffffffu);         // global_addr low
  g0[3] = (unsigned)((ga >> 32) & 0x01ffffffu)  // global_addr[56:32]
          | (2u << 30);                         // type = 2 ("image")
  v8i g1;
  g1[0] = (int)((1u << 16)    // data_size = 2 bytes
                | (1u << 20)  // pad_enable
                | (6u << 22)  // pad_interval: 128 dwords (512B)
                | (3u << 25));// pad_amount: 4 dwords (16B)
  g1[1] = (int)(256u << 16);  // tensor_dim0 = 256        (bits 79:48)
  g1[2] = (int)(32u << 16);   // tensor_dim1 = 32         (bits 111:80)
  g1[3] = (int)(256u << 16);  // tile_dim0 = 256          (bits 127:112)
  g1[4] = 32;                 // tile_dim1 = 32, tile_dim2 = 0
  g1[5] = 256;                // tensor_dim0_stride = 256 (bits 207:160)
  g1[6] = 0;
  g1[7] = 0;
  const v4i gz4 = {0, 0, 0, 0};                  // groups 2/3 unused (2-D)
  const v8i gz8 = {0, 0, 0, 0, 0, 0, 0, 0};
  // 6-arg form (amdgpu-toolchain / clang-23 signature)
  __builtin_amdgcn_tensor_load_to_lds(g0, g1, gz4, gz4, gz8, 0);
}
#endif

static __device__ __forceinline__ void wait_tensor0() {
#if HAVE_TCNT
  __builtin_amdgcn_s_wait_tensorcnt(0);
#elif HAVE_TDM
  asm volatile("s_wait_tensorcnt 0x0" ::: "memory");
#endif
}

// ---------------------------------------------------------------------------
// Kernel 1: convert W [in,out] f32 -> Wb [z][out][in] bf16 (transposed)
// ---------------------------------------------------------------------------
__global__ __launch_bounds__(256) void wconv_kernel(
    const float* __restrict__ Wq, const float* __restrict__ Wk,
    const float* __restrict__ Wv, __bf16* __restrict__ Wb) {
  const int t = blockIdx.x * 256 + threadIdx.x; // 0 .. 3*256*256-1
  const int z = t >> 16;
  const int n = (t >> 8) & 255;
  const int k = t & 255;
  const float* W = (z == 0) ? Wq : (z == 1) ? Wk : Wv;
  Wb[t] = (__bf16)W[k * CC + n];
}

// ---------------------------------------------------------------------------
// Kernel 2: QKV projection. block = 64 rows of x, 8 waves, wave -> 16x128 tile
// q additionally scaled by 1/sqrt(C) = 1/16 at store.
// ---------------------------------------------------------------------------
__global__ __launch_bounds__(256) void qkv_proj_kernel(
    const float* __restrict__ x, const float* __restrict__ bq,
    const float* __restrict__ bk, const float* __restrict__ bv,
    const __bf16* __restrict__ Wb, __bf16* __restrict__ qkv) {
  __shared__ __align__(16) __bf16 Xs[64][264]; // padded: no bank conflicts
  const int tid = threadIdx.x;
  const int z = blockIdx.z;
  const int m0 = blockIdx.x * 64;

#pragma unroll
  for (int i = 0; i < 16; ++i) {
    const int chunk = tid + i * 256; // 4096 float4 chunks
    const int row = chunk >> 6;
    const int c4 = (chunk & 63) << 2;
    const float4 d = *(const float4*)(x + (size_t)(m0 + row) * CC + c4);
    Xs[row][c4 + 0] = (__bf16)d.x;
    Xs[row][c4 + 1] = (__bf16)d.y;
    Xs[row][c4 + 2] = (__bf16)d.z;
    Xs[row][c4 + 3] = (__bf16)d.w;
  }
  __syncthreads();

  const int w = tid >> 5, l = tid & 31, half = l >> 4, ln = l & 15;
  const int msub = w & 3;
  const int nh = w >> 2;
  const __bf16* Wz = Wb + (size_t)z * CC * CC;

  const v8f vzero = {0.f, 0.f, 0.f, 0.f, 0.f, 0.f, 0.f, 0.f};
  v8f acc[8];
#pragma unroll
  for (int j = 0; j < 8; ++j) acc[j] = vzero;

#pragma unroll
  for (int ks = 0; ks < 8; ++ks) {
    const int k0 = ks * 32;
    const __bf16* ap = &Xs[msub * 16 + ln][k0 + 8 * half];
    const v16bf a = mk16(*(const v8bf*)ap, *(const v8bf*)(ap + 16));
#pragma unroll
    for (int j = 0; j < 8; ++j) {
      const int n = (nh * 8 + j) * 16 + ln;
      const __bf16* bp = Wz + (size_t)n * CC + k0 + 8 * half;
      const v16bf bfr = mk16(*(const v8bf*)bp, *(const v8bf*)(bp + 16));
      acc[j] = wmma_bf16(a, bfr, acc[j]);
    }
  }

  const float* bias = (z == 0) ? bq : (z == 1) ? bk : bv;
  const float osc = (z == 0) ? 0.0625f : 1.0f;
  __bf16* dst = qkv + (size_t)z * MTOT * CC;
#pragma unroll
  for (int j = 0; j < 8; ++j) {
    const int col = (nh * 8 + j) * 16 + ln;
    const float bbv = bias[col];
#pragma unroll
    for (int r = 0; r < 8; ++r) {
      const int row = m0 + msub * 16 + 8 * half + r;
      dst[(size_t)row * CC + col] = (__bf16)((acc[j][r] + bbv) * osc);
    }
  }
}

// ---------------------------------------------------------------------------
// Kernel 3: causal flash attention. block = 128 query rows (8 waves x 16).
// K tiles: double-buffered in LDS, DMA'd by the Tensor Data Mover (wave 0
// issues next tile's tensor_load_to_lds while everyone computes the current
// one; s_wait_tensorcnt + barrier publishes it). V tiles: manual transpose.
// ---------------------------------------------------------------------------
__global__ __launch_bounds__(256) void attn_kernel(
    const __bf16* __restrict__ Qb, const __bf16* __restrict__ Kb,
    const __bf16* __restrict__ Vb, float* __restrict__ out) {
  __shared__ __align__(16) __bf16 Ks[2][32][264]; // double-buffered K tile
  __shared__ __align__(16) __bf16 Vt[256][40];    // V tile transposed [c][s]
  __shared__ __align__(16) __bf16 Pb[8][16][40];  // per-wave P relayout buf

  const int tid = threadIdx.x;
  const int w = tid >> 5, l = tid & 31, half = l >> 4, ln = l & 15;
  const int bq = blockIdx.y;
  const int qblk = blockIdx.x;
  const __bf16* Q = Qb + (size_t)bq * TT * CC;
  const __bf16* K = Kb + (size_t)bq * TT * CC;
  const __bf16* V = Vb + (size_t)bq * TT * CC;
  const int qrow0 = qblk * 128 + w * 16;

  const v8f vzero = {0.f, 0.f, 0.f, 0.f, 0.f, 0.f, 0.f, 0.f};
  v8f acc[16];
#pragma unroll
  for (int ct = 0; ct < 16; ++ct) acc[ct] = vzero;
  float mi[8], li[8];
#pragma unroll
  for (int r = 0; r < 8; ++r) { mi[r] = -3.0e38f; li[r] = 0.f; }

  const int nkb = (qblk + 1) * 4; // uniform across block -> barriers legal

  // Prologue: stage K tile 0 into buffer 0.
#if HAVE_TDM
  if (w == 0) {
    tdm_load_k_tile(K, lds_off(&Ks[0][0][0]));
    asm volatile("" ::: "memory");
  }
#else
#pragma unroll
  for (int i = 0; i < 4; ++i) {
    const int chunk = tid + i * 256;
    const int row = chunk >> 5;
    const int c8 = (chunk & 31) << 3;
    *(v8bf*)&Ks[0][row][c8] = *(const v8bf*)(K + (size_t)row * CC + c8);
  }
#endif

  for (int kb = 0; kb < nkb; ++kb) {
    const int s0 = kb * 32;
    const int cur = kb & 1;
#if HAVE_TDM
    if (w == 0) wait_tensor0(); // K tile kb landed in LDS (wave 0's DMA)
#endif
    __syncthreads(); // publish K tile kb; prior iteration's LDS reads done

#if HAVE_TDM
    if (w == 0 && kb + 1 < nkb) { // overlap next K DMA with compute
      tdm_load_k_tile(K + (size_t)(s0 + 32) * CC, lds_off(&Ks[cur ^ 1][0][0]));
      asm volatile("" ::: "memory");
    }
#else
    if (kb + 1 < nkb) {
#pragma unroll
      for (int i = 0; i < 4; ++i) {
        const int chunk = tid + i * 256;
        const int row = chunk >> 5;
        const int c8 = (chunk & 31) << 3;
        *(v8bf*)&Ks[cur ^ 1][row][c8] =
            *(const v8bf*)(K + (size_t)(s0 + 32 + row) * CC + c8);
      }
    }
#endif

    // V tile (manual transposed store into Vt)
#pragma unroll
    for (int i = 0; i < 4; ++i) {
      const int chunk = tid + i * 256; // 1024 8-elem chunks (32 rows x 256)
      const int row = chunk >> 5;
      const int c8 = (chunk & 31) << 3;
      const v8bf dv = *(const v8bf*)(V + (size_t)(s0 + row) * CC + c8);
#pragma unroll
      for (int j = 0; j < 8; ++j) Vt[c8 + j][row] = dv[j];
      if (kb + 1 < nkb)
        __builtin_prefetch(V + (size_t)(s0 + 32 + row) * CC + c8, 0, 1);
    }
    __syncthreads();

    // S = Q * K^T for 16 rows x 32 keys (two 16x16 n-tiles)
    v8f sc0 = vzero, sc1 = vzero;
#pragma unroll
    for (int ks = 0; ks < 8; ++ks) {
      const int k0 = ks * 32;
      const __bf16* qp = Q + (size_t)(qrow0 + ln) * CC + k0 + 8 * half;
      const v16bf a = mk16(*(const v8bf*)qp, *(const v8bf*)(qp + 16));
      const __bf16* kp0 = &Ks[cur][ln][k0 + 8 * half];
      const v16bf b0 = mk16(*(const v8bf*)kp0, *(const v8bf*)(kp0 + 16));
      sc0 = wmma_bf16(a, b0, sc0);
      const __bf16* kp1 = &Ks[cur][16 + ln][k0 + 8 * half];
      const v16bf b1 = mk16(*(const v8bf*)kp1, *(const v8bf*)(kp1 + 16));
      sc1 = wmma_bf16(a, b1, sc1);
    }

    // causal mask + online softmax (row lives across a 16-lane half)
#pragma unroll
    for (int r = 0; r < 8; ++r) {
      const int t = qrow0 + 8 * half + r;
      float v0 = sc0[r]; if (s0 + ln      > t) v0 = -3.0e38f;
      float v1 = sc1[r]; if (s0 + 16 + ln > t) v1 = -3.0e38f;
      const float rm = redmax16(fmaxf(v0, v1));
      const float mn = fmaxf(mi[r], rm);
      const float sc = __expf(mi[r] - mn);
      const float p0 = __expf(v0 - mn);
      const float p1 = __expf(v1 - mn);
      const float ls = redsum16(p0 + p1);
      li[r] = li[r] * sc + ls;
      mi[r] = mn;
#pragma unroll
      for (int ct = 0; ct < 16; ++ct) acc[ct][r] *= sc;
      Pb[w][8 * half + r][ln]      = (__bf16)p0;
      Pb[w][8 * half + r][16 + ln] = (__bf16)p1;
    }
    __syncthreads(); // Pb visible wave-wide (uniform barrier)

    // O += P(16x32) * V(32x256)
    const __bf16* pp = &Pb[w][ln][8 * half];
    const v16bf ap = mk16(*(const v8bf*)pp,
                          *(const v8bf*)(&Pb[w][ln][16 + 8 * half]));
#pragma unroll
    for (int ct = 0; ct < 16; ++ct) {
      const __bf16* vp = &Vt[ct * 16 + ln][8 * half];
      const v16bf bv = mk16(*(const v8bf*)vp, *(const v8bf*)(vp + 16));
      acc[ct] = wmma_bf16(ap, bv, acc[ct]);
    }
  }

  // epilogue: divide by softmax denominator, store f32
#pragma unroll
  for (int ct = 0; ct < 16; ++ct) {
#pragma unroll
    for (int r = 0; r < 8; ++r) {
      const int t = qrow0 + 8 * half + r;
      out[((size_t)bq * TT + t) * CC + ct * 16 + ln] = acc[ct][r] / li[r];
    }
  }
}

// ---------------------------------------------------------------------------
extern "C" void kernel_launch(void* const* d_in, const int* in_sizes, int n_in,
                              void* d_out, int out_size, void* d_ws,
                              size_t ws_size, hipStream_t stream) {
  (void)in_sizes; (void)n_in; (void)out_size; (void)ws_size;
  const float* x  = (const float*)d_in[0];
  const float* Wq = (const float*)d_in[1];
  const float* bq = (const float*)d_in[2];
  const float* Wk = (const float*)d_in[3];
  const float* bk = (const float*)d_in[4];
  const float* Wv = (const float*)d_in[5];
  const float* bv = (const float*)d_in[6];
  float* out = (float*)d_out;

  // workspace: [Q | K | V] bf16 (3 * 16384*256) then Wb bf16 (3 * 256*256)
  __bf16* qkv = (__bf16*)d_ws;
  __bf16* Wb  = qkv + (size_t)3 * MTOT * CC;

  wconv_kernel<<<dim3(768), dim3(256), 0, stream>>>(Wq, Wk, Wv, Wb);
  qkv_proj_kernel<<<dim3(256, 1, 3), dim3(256), 0, stream>>>(x, bq, bk, bv,
                                                             Wb, qkv);
  attn_kernel<<<dim3(32, 4), dim3(256), 0, stream>>>(
      qkv, qkv + (size_t)MTOT * CC, qkv + (size_t)2 * MTOT * CC, out);
}
